// JointAttention_37984690766218
// MI455X (gfx1250) — matrix operand order repaired and checked by reading, not compiled, not run on hardware
//
#include <hip/hip_runtime.h>

typedef __bf16  v16bf __attribute__((ext_vector_type(16)));
typedef float   v8f   __attribute__((ext_vector_type(8)));

union FragBF { v16bf v; unsigned int u[8]; };

__device__ __forceinline__ unsigned short f2bf(float f) {
    unsigned int u = __float_as_uint(f);
    u = (u + 0x7FFFu + ((u >> 16) & 1u)) >> 16;
    return (unsigned short)u;
}

__device__ __forceinline__ v8f wmma_bf16(v16bf a, v16bf b, v8f c) {
    return __builtin_amdgcn_wmma_f32_16x16x32_bf16(false, a, false, b, (short)0, c, false, false);
}

// async copy of 32 contiguous bytes global -> LDS (two b128), ASYNCcnt-tracked
__device__ __forceinline__ void async_copy_b256(unsigned lds_off, const void* gptr) {
    asm volatile(
        "global_load_async_to_lds_b128 %0, %1, off\n\t"
        "global_load_async_to_lds_b128 %0, %1, off offset:16"
        :: "v"(lds_off), "v"(gptr) : "memory");
}
__device__ __forceinline__ void wait_async0() {
    asm volatile("s_wait_asynccnt 0x0" ::: "memory");
}

// A-fragment (16x32 bf16, M x K): lane m = lane&15, kh = (lane>>4)*8,
// elements 0..7 -> K = coff+kh..+7 ; elements 8..15 -> K = coff+16+kh..+7
__device__ __forceinline__ v16bf load_afrag(const unsigned short* base, int stride, int coff, int lane) {
    int m  = lane & 15;
    int kh = (lane >> 4) << 3;
    const unsigned int* p0 = (const unsigned int*)(base + (size_t)m * stride + coff + kh);
    const unsigned int* p1 = (const unsigned int*)(base + (size_t)m * stride + coff + 16 + kh);
    FragBF r;
#pragma unroll
    for (int i = 0; i < 4; ++i) { r.u[i] = p0[i]; r.u[4 + i] = p1[i]; }
    return r.v;
}

// B-fragment (32x16 bf16, K x N): lane n = lane&15, kh = (lane>>4)*8,
// elements 0..7 -> K = kh..+7 ; elements 8..15 -> K = 16+kh..+7. base must point at row n=0, K contiguous.
__device__ __forceinline__ v16bf load_bfrag(const unsigned short* base, int stride, int lane) {
    int n  = lane & 15;
    int kh = (lane >> 4) << 3;
    const unsigned int* p = (const unsigned int*)(base + (size_t)n * stride + kh);
    FragBF r;
#pragma unroll
    for (int i = 0; i < 4; ++i) { r.u[i] = p[i]; r.u[4 + i] = p[8 + i]; }
    return r.v;
}

__device__ __forceinline__ float rmax16(float v) {
    v = fmaxf(v, __shfl_xor(v, 1, 32));
    v = fmaxf(v, __shfl_xor(v, 2, 32));
    v = fmaxf(v, __shfl_xor(v, 4, 32));
    v = fmaxf(v, __shfl_xor(v, 8, 32));
    return v;
}
__device__ __forceinline__ float rsum16(float v) {
    v += __shfl_xor(v, 1, 32);
    v += __shfl_xor(v, 2, 32);
    v += __shfl_xor(v, 4, 32);
    v += __shfl_xor(v, 8, 32);
    return v;
}
__device__ __forceinline__ float rsum32(float v) {
    v += __shfl_xor(v, 1, 32);
    v += __shfl_xor(v, 2, 32);
    v += __shfl_xor(v, 4, 32);
    v += __shfl_xor(v, 8, 32);
    v += __shfl_xor(v, 16, 32);
    return v;
}

// ---------------------------------------------------------------- convert
__global__ void cvt_f32_bf16(const float* __restrict__ in, unsigned short* __restrict__ out, int n) {
    int i = blockIdx.x * blockDim.x + threadIdx.x;
    int stride = gridDim.x * blockDim.x;
    for (; i < n; i += stride) out[i] = f2bf(in[i]);
}

// ---------------------------------------------------------------- generic WMMA GEMM
// C(MxN, f32, row-major) = A(MxK, bf16, row-major) * B(KxN, bf16, row-major)
// block = 128 threads (4 waves). Tile: 64x64 C, K-chunks of 32.
// Requires M%64==0, N%64==0, K%32==0.
__global__ void gemm_bf16_wmma(const unsigned short* __restrict__ A,
                               const unsigned short* __restrict__ B,
                               float* __restrict__ C,
                               int M, int N, int K) {
    __shared__ unsigned short As[64][40];   // 64 rows x 32 K (+8 pad)
    __shared__ unsigned short BsT[64][40];  // 64 N rows x 32 K (+8 pad), transposed

    const int tid  = threadIdx.x;
    const int lane = tid & 31;
    const int w    = tid >> 5;
    const int mbase = blockIdx.y * 64;
    const int nbase = blockIdx.x * 64;

    v8f acc[4];
#pragma unroll
    for (int nb = 0; nb < 4; ++nb)
#pragma unroll
        for (int r = 0; r < 8; ++r) acc[nb][r] = 0.0f;

    const int arow = tid >> 1, aseg = (tid & 1) * 16;   // A: 64 rows x (2 segs of 16)
    const int bk   = tid >> 2, bseg = (tid & 3) * 16;   // B: 32 k-rows x (4 segs of 16)
    const unsigned as_off = (unsigned)(size_t)&As[arow][aseg];

    for (int kk = 0; kk < K; kk += 32) {
        __syncthreads();
        // stage A tile via async global->LDS DMA (ASYNCcnt)
        async_copy_b256(as_off, (const void*)(A + (size_t)(mbase + arow) * K + kk + aseg));
        // stage B tile transposed (BsT[n][k] = B[kk+k][nbase+n])
        union { uint4 q[2]; unsigned short us[16]; } bb;
        const uint4* gb = (const uint4*)(B + (size_t)(kk + bk) * N + nbase + bseg);
        bb.q[0] = gb[0]; bb.q[1] = gb[1];
#pragma unroll
        for (int i = 0; i < 16; ++i) BsT[bseg + i][bk] = bb.us[i];

        if (kk + 32 < K)
            __builtin_prefetch((const void*)(B + (size_t)(kk + 32 + bk) * N + nbase + bseg), 0, 1);

        wait_async0();
        __syncthreads();

        v16bf af = load_afrag(&As[w * 16][0], 40, 0, lane);
        v16bf b0 = load_bfrag(&BsT[0][0],  40, lane);
        v16bf b1 = load_bfrag(&BsT[16][0], 40, lane);
        v16bf b2 = load_bfrag(&BsT[32][0], 40, lane);
        v16bf b3 = load_bfrag(&BsT[48][0], 40, lane);
        acc[0] = wmma_bf16(af, b0, acc[0]);
        acc[1] = wmma_bf16(af, b1, acc[1]);
        acc[2] = wmma_bf16(af, b2, acc[2]);
        acc[3] = wmma_bf16(af, b3, acc[3]);
    }

    const int n = lane & 15, half = lane >> 4;
#pragma unroll
    for (int nb = 0; nb < 4; ++nb)
#pragma unroll
        for (int r = 0; r < 8; ++r)
            C[(size_t)(mbase + w * 16 + r + 8 * half) * N + nbase + nb * 16 + n] = acc[nb][r];
}

// ---------------------------------------------------------------- RMSNorm + RoPE + pack
// grid = (2304, 24), block = 32. Lane l owns dims l and l+32 (rotate-half pair in-lane).
__global__ void normrope_pack(const float* __restrict__ qkv0, const float* __restrict__ qkv1,
                              const float* __restrict__ gq0, const float* __restrict__ gk0,
                              const float* __restrict__ gq1, const float* __restrict__ gk1,
                              unsigned short* __restrict__ Qb, unsigned short* __restrict__ Kb,
                              unsigned short* __restrict__ Vb) {
    const int S = 2304, S0 = 2048;
    const int sg = blockIdx.x;
    const int h  = blockIdx.y;
    const int l  = threadIdx.x;

    const int mod = (sg >= S0) ? 1 : 0;
    const int sl  = mod ? (sg - S0) : sg;
    const float* row = mod ? (qkv1 + (size_t)sl * 4608) : (qkv0 + (size_t)sl * 4608);
    const float* gq  = mod ? gq1 : gq0;
    const float* gk  = mod ? gk1 : gk0;

    float qa = row[h * 64 + l],        qb_ = row[h * 64 + l + 32];
    float ka = row[1536 + h * 64 + l], kb_ = row[1536 + h * 64 + l + 32];
    float va = row[3072 + h * 64 + l], vb_ = row[3072 + h * 64 + l + 32];

    float sq = rsum32(qa * qa + qb_ * qb_);
    float sk = rsum32(ka * ka + kb_ * kb_);
    float rq = rsqrtf(sq * (1.0f / 64.0f) + 1e-6f);
    float rk = rsqrtf(sk * (1.0f / 64.0f) + 1e-6f);
    qa *= rq * gq[l]; qb_ *= rq * gq[l + 32];
    ka *= rk * gk[l]; kb_ *= rk * gk[l + 32];

    // RoPE: freq index j = l, inv = 10000^(-2l/64)
    float inv = __expf(-((2.0f * (float)l) / 64.0f) * 9.210340371976184f); // ln(10000)
    float f   = (float)sl * inv;
    float cs = __cosf(f), sn = __sinf(f);
    float q0 = qa * cs - qb_ * sn, q1 = qb_ * cs + qa * sn;
    float k0 = ka * cs - kb_ * sn, k1 = kb_ * cs + ka * sn;

    size_t base = ((size_t)h * S + sg) * 64;
    Qb[base + l] = f2bf(q0); Qb[base + l + 32] = f2bf(q1);
    Kb[base + l] = f2bf(k0); Kb[base + l + 32] = f2bf(k1);
    Vb[base + l] = f2bf(va); Vb[base + l + 32] = f2bf(vb_);
}

// ---------------------------------------------------------------- flash attention
// grid = (S/64, H), block = 128 (4 waves). Each wave: 16 queries, online softmax,
// key blocks of 32 staged in LDS (K via async DMA, V transposed). AO: (S, H*64) bf16.
__global__ void attn_fa_wmma(const unsigned short* __restrict__ Q,
                             const unsigned short* __restrict__ K,
                             const unsigned short* __restrict__ V,
                             unsigned short* __restrict__ AO) {
    const int S = 2304;
    __shared__ unsigned short Ks[32][80];      // 32 keys x 64 dims (+16 pad)
    __shared__ unsigned short Vt[64][40];      // 64 dims x 32 keys (+8 pad) (transposed)
    __shared__ unsigned short Ps[4][16][40];   // per-wave P tile 16x32 (+8 pad)

    const int tid  = threadIdx.x;
    const int lane = tid & 31;
    const int w    = tid >> 5;
    const int h    = blockIdx.y;
    const int qbase = blockIdx.x * 64 + w * 16;

    // Q fragments held in registers for the whole loop (global is row-major, K contiguous)
    const unsigned short* Qrow = Q + ((size_t)h * S + qbase) * 64;
    v16bf aq0 = load_afrag(Qrow, 64, 0, lane);
    v16bf aq1 = load_afrag(Qrow, 64, 32, lane);

    float mrow[8], lrow[8];
    v8f o[4];
#pragma unroll
    for (int r = 0; r < 8; ++r) { mrow[r] = -1e30f; lrow[r] = 0.0f; }
#pragma unroll
    for (int nb = 0; nb < 4; ++nb)
#pragma unroll
        for (int r = 0; r < 8; ++r) o[nb][r] = 0.0f;

    const int krow = tid >> 2, kseg = (tid & 3) * 16;
    const unsigned ks_off = (unsigned)(size_t)&Ks[krow][kseg];
    const float scale = 0.125f;           // 1/sqrt(64)
    const float L2E   = 1.44269504088896f;

    for (int kb = 0; kb < S; kb += 32) {
        __syncthreads();
        // stage K block row-major via async global->LDS DMA (ASYNCcnt)
        async_copy_b256(ks_off, (const void*)(K + ((size_t)h * S + kb + krow) * 64 + kseg));
        // stage V block transposed (Vt[d][k])
        union { uint4 q[2]; unsigned short us[16]; } vv;
        const uint4* gv = (const uint4*)(V + ((size_t)h * S + kb + krow) * 64 + kseg);
        vv.q[0] = gv[0]; vv.q[1] = gv[1];
#pragma unroll
        for (int i = 0; i < 16; ++i) Vt[kseg + i][krow] = vv.us[i];
        wait_async0();
        __syncthreads();

        // scores: two 16x16 tiles (key groups 0..15, 16..31), K-dim 64 = 2 chunks
        v8f s0, s1;
#pragma unroll
        for (int r = 0; r < 8; ++r) { s0[r] = 0.0f; s1[r] = 0.0f; }
        v16bf bk00 = load_bfrag(&Ks[0][0],   80, lane);
        v16bf bk01 = load_bfrag(&Ks[0][32],  80, lane);
        v16bf bk10 = load_bfrag(&Ks[16][0],  80, lane);
        v16bf bk11 = load_bfrag(&Ks[16][32], 80, lane);
        s0 = wmma_bf16(aq0, bk00, s0);
        s0 = wmma_bf16(aq1, bk01, s0);
        s1 = wmma_bf16(aq0, bk10, s1);
        s1 = wmma_bf16(aq1, bk11, s1);

        // online softmax (row lives across 16 lanes of this half-wave)
        float p0[8], p1[8];
#pragma unroll
        for (int r = 0; r < 8; ++r) {
            float a = s0[r] * scale, b = s1[r] * scale;
            float mb = rmax16(fmaxf(a, b));
            float mn = fmaxf(mrow[r], mb);
            float alpha = exp2f((mrow[r] - mn) * L2E);
            float e0 = exp2f((a - mn) * L2E);
            float e1 = exp2f((b - mn) * L2E);
            lrow[r] = lrow[r] * alpha + rsum16(e0 + e1);
            mrow[r] = mn;
            p0[r] = e0; p1[r] = e1;
#pragma unroll
            for (int nb = 0; nb < 4; ++nb) o[nb][r] *= alpha;
        }

        // re-stripe P (C/D layout) into A-fragment layout via per-wave LDS
        const int n = lane & 15, half = lane >> 4;
#pragma unroll
        for (int r = 0; r < 8; ++r) {
            int M = r + 8 * half;
            Ps[w][M][n]      = f2bf(p0[r]);
            Ps[w][M][16 + n] = f2bf(p1[r]);
        }
        v16bf pa = load_afrag(&Ps[w][0][0], 40, 0, lane);
        v16bf bv0 = load_bfrag(&Vt[0][0],  40, lane);
        v16bf bv1 = load_bfrag(&Vt[16][0], 40, lane);
        v16bf bv2 = load_bfrag(&Vt[32][0], 40, lane);
        v16bf bv3 = load_bfrag(&Vt[48][0], 40, lane);
        o[0] = wmma_bf16(pa, bv0, o[0]);
        o[1] = wmma_bf16(pa, bv1, o[1]);
        o[2] = wmma_bf16(pa, bv2, o[2]);
        o[3] = wmma_bf16(pa, bv3, o[3]);
    }

    // normalize and write packed (S, H*64) bf16
    const int n = lane & 15, half = lane >> 4;
#pragma unroll
    for (int r = 0; r < 8; ++r) {
        float inv = 1.0f / lrow[r];
        int sg = qbase + r + 8 * half;
#pragma unroll
        for (int nb = 0; nb < 4; ++nb)
            AO[(size_t)sg * 1536 + h * 64 + nb * 16 + n] = f2bf(o[nb][r] * inv);
    }
}

// ---------------------------------------------------------------- launch
extern "C" void kernel_launch(void* const* d_in, const int* in_sizes, int n_in,
                              void* d_out, int out_size, void* d_ws, size_t ws_size,
                              hipStream_t stream) {
    const float* x0    = (const float*)d_in[0];
    const float* x1    = (const float*)d_in[1];
    const float* wqkv0 = (const float*)d_in[2];
    const float* wqkv1 = (const float*)d_in[3];
    const float* wout0 = (const float*)d_in[4];
    const float* wout1 = (const float*)d_in[5];
    const float* gq0   = (const float*)d_in[6];
    const float* gk0   = (const float*)d_in[7];
    const float* gq1   = (const float*)d_in[8];
    const float* gk1   = (const float*)d_in[9];
    float* out = (float*)d_out;

    const int S0 = 2048, S1 = 256, S = 2304, DIM = 1536, INNER = 1536, NQKV = 4608;
    const int H = 24;

    char* ws = (char*)d_ws;
    size_t off = 0;
    auto alloc = [&](size_t bytes) -> char* {
        char* p = ws + off;
        off += (bytes + 255) & ~(size_t)255;
        return p;
    };

    unsigned short* x0b  = (unsigned short*)alloc((size_t)S0 * DIM * 2);
    unsigned short* x1b  = (unsigned short*)alloc((size_t)S1 * DIM * 2);
    unsigned short* wq0b = (unsigned short*)alloc((size_t)DIM * NQKV * 2);
    unsigned short* wq1b = (unsigned short*)alloc((size_t)DIM * NQKV * 2);
    unsigned short* wo0b = (unsigned short*)alloc((size_t)INNER * DIM * 2);
    unsigned short* wo1b = (unsigned short*)alloc((size_t)INNER * DIM * 2);
    float* qkv0 = (float*)alloc((size_t)S0 * NQKV * 4);
    float* qkv1 = (float*)alloc((size_t)S1 * NQKV * 4);
    unsigned short* Qb = (unsigned short*)alloc((size_t)H * S * 64 * 2);
    unsigned short* Kb = (unsigned short*)alloc((size_t)H * S * 64 * 2);
    unsigned short* Vb = (unsigned short*)alloc((size_t)H * S * 64 * 2);
    unsigned short* AO = (unsigned short*)alloc((size_t)S * INNER * 2);

    auto cvt = [&](const float* src, unsigned short* dst, int n) {
        int blocks = (n + 1023) / 1024;
        cvt_f32_bf16<<<dim3(blocks), dim3(1024), 0, stream>>>(src, dst, n);
    };
    cvt(x0, x0b, S0 * DIM);
    cvt(x1, x1b, S1 * DIM);
    cvt(wqkv0, wq0b, DIM * NQKV);
    cvt(wqkv1, wq1b, DIM * NQKV);
    cvt(wout0, wo0b, INNER * DIM);
    cvt(wout1, wo1b, INNER * DIM);

    // QKV projections
    gemm_bf16_wmma<<<dim3(NQKV / 64, S0 / 64), dim3(128), 0, stream>>>(x0b, wq0b, qkv0, S0, NQKV, DIM);
    gemm_bf16_wmma<<<dim3(NQKV / 64, S1 / 64), dim3(128), 0, stream>>>(x1b, wq1b, qkv1, S1, NQKV, DIM);

    // RMSNorm + RoPE + pack to (h, S, d) bf16
    normrope_pack<<<dim3(S, H), dim3(32), 0, stream>>>(qkv0, qkv1, gq0, gk0, gq1, gk1, Qb, Kb, Vb);

    // joint attention
    attn_fa_wmma<<<dim3(S / 64, H), dim3(128), 0, stream>>>(Qb, Kb, Vb, AO);

    // output projections straight into d_out (o0 then o1, both f32)
    gemm_bf16_wmma<<<dim3(DIM / 64, S0 / 64), dim3(128), 0, stream>>>(AO, wo0b, out, S0, DIM, INNER);
    gemm_bf16_wmma<<<dim3(DIM / 64, S1 / 64), dim3(128), 0, stream>>>(
        AO + (size_t)S0 * INNER, wo1b, out + (size_t)S0 * DIM, S1, DIM, INNER);
}
